// Attention_21603685499660
// MI455X (gfx1250) — compile-verified, matrix operand order checked by reading, not run on hardware
//
#include <hip/hip_runtime.h>

typedef __attribute__((ext_vector_type(16))) __bf16 v16bf;
typedef __attribute__((ext_vector_type(8)))  float  v8f;
typedef __attribute__((ext_vector_type(4)))  unsigned v4u;
typedef __attribute__((ext_vector_type(8)))  int    v8i;
typedef __attribute__((ext_vector_type(4)))  int    v4i;

#if defined(__has_builtin)
#if __has_builtin(__builtin_amdgcn_tensor_load_to_lds) && __has_builtin(__builtin_amdgcn_s_wait_tensorcnt)
#define HAVE_TDM 1
#endif
#if __has_builtin(__builtin_amdgcn_cvt_pk_bf16_f32)
#define HAVE_CVT_PK_BF16 1
#endif
#endif
#ifndef HAVE_TDM
#define HAVE_TDM 0
#endif
#ifndef HAVE_CVT_PK_BF16
#define HAVE_CVT_PK_BF16 0
#endif

template<bool B> struct BoolC { static constexpr bool value = B; };

union FragU { uint4 q[2]; v16bf v; };

__device__ __forceinline__ unsigned short f2bf(float f) {
    unsigned u = __float_as_uint(f);
    u += 0x7FFFu + ((u >> 16) & 1u);   // round-to-nearest-even
    return (unsigned short)(u >> 16);
}
__device__ __forceinline__ unsigned pk2(float a, float b) {
#if HAVE_CVT_PK_BF16
    auto r = __builtin_amdgcn_cvt_pk_bf16_f32(a, b);   // v_cvt_pk_bf16_f32 (RNE)
    static_assert(sizeof(r) == 4, "cvt_pk_bf16 size");
    unsigned u;
    __builtin_memcpy(&u, &r, 4);
    return u;
#else
    return (unsigned)f2bf(a) | ((unsigned)f2bf(b) << 16);
#endif
}
__device__ __forceinline__ float bf2f(unsigned short h) {
    return __uint_as_float(((unsigned)h) << 16);
}

// ---------------------------------------------------------------------------
// C[M,N] = A[M,K] * W[N,K]^T   (both operands K-contiguous -> identical
// A-frag / B-frag lane addressing for V_WMMA_F32_16X16X32_BF16)
// Block: 256 thr = 8 waves (4 M x 2 N), tile 128x128, BK=32, double-buffered LDS
// ---------------------------------------------------------------------------
template<bool A_BF16, bool OUT_BF16>
__global__ __launch_bounds__(256) void gemm_nt_bf16(
    const void* __restrict__ Ap, const float* __restrict__ W,
    void* __restrict__ Cp, int M, int N, int K)
{
    __shared__ unsigned ldsA[2 * 128 * 24];   // 128 rows x (32+16 pad) bf16, x2 buffers
    __shared__ unsigned ldsB[2 * 128 * 24];

    const int tid  = threadIdx.x;
    const int lane = tid & 31, wave = tid >> 5;
    const int llo  = lane & 15, lhi = lane >> 4;
    const int waveM = wave >> 1, waveN = wave & 1;
    const int m0 = blockIdx.y * 128, n0 = blockIdx.x * 128;
    const int r = tid >> 1, half = tid & 1;           // LDS-fill mapping

    v8f vzero = {};
    v8f acc[2][4];
    #pragma unroll
    for (int mt = 0; mt < 2; ++mt)
        #pragma unroll
        for (int nt = 0; nt < 4; ++nt) acc[mt][nt] = vzero;

    // running global pointers for this thread's staging slice
    const float4* gaF = (const float4*)((const float*)Ap + (size_t)(m0 + r) * K + half * 16);
    const uint4*  gaH = (const uint4*)((const unsigned short*)Ap + (size_t)(m0 + r) * K + half * 16);
    const float4* gw  = (const float4*)(W + (size_t)(n0 + r) * K + half * 16);

    float4 ra[4]; uint4 rh[2]; float4 rw[4];

    // ---- prime: load k-step 0 into regs ----
    if (A_BF16) { rh[0] = gaH[0]; rh[1] = gaH[1]; }
    else        { ra[0] = gaF[0]; ra[1] = gaF[1]; ra[2] = gaF[2]; ra[3] = gaF[3]; }
    rw[0] = gw[0]; rw[1] = gw[1]; rw[2] = gw[2]; rw[3] = gw[3];

    // ---- store k-step 0 into buffer 0 ----
    {
        if (A_BF16) {
            uint4* d = (uint4*)(ldsA + r * 24 + half * 8);
            d[0] = rh[0]; d[1] = rh[1];
        } else {
            unsigned* d = ldsA + r * 24 + half * 8;
            d[0] = pk2(ra[0].x, ra[0].y); d[1] = pk2(ra[0].z, ra[0].w);
            d[2] = pk2(ra[1].x, ra[1].y); d[3] = pk2(ra[1].z, ra[1].w);
            d[4] = pk2(ra[2].x, ra[2].y); d[5] = pk2(ra[2].z, ra[2].w);
            d[6] = pk2(ra[3].x, ra[3].y); d[7] = pk2(ra[3].z, ra[3].w);
        }
        unsigned* d = ldsB + r * 24 + half * 8;
        d[0] = pk2(rw[0].x, rw[0].y); d[1] = pk2(rw[0].z, rw[0].w);
        d[2] = pk2(rw[1].x, rw[1].y); d[3] = pk2(rw[1].z, rw[1].w);
        d[4] = pk2(rw[2].x, rw[2].y); d[5] = pk2(rw[2].z, rw[2].w);
        d[6] = pk2(rw[3].x, rw[3].y); d[7] = pk2(rw[3].z, rw[3].w);
    }
    __syncthreads();

    const int nsteps = K >> 5;
    for (int s = 0; s < nsteps; ++s) {
        const bool more = (s + 1) < nsteps;
        unsigned* curA = ldsA + (s & 1) * (128 * 24);
        unsigned* curB = ldsB + (s & 1) * (128 * 24);

        // ---- prefetch next k-step into regs (overlaps with WMMA below) ----
        if (more) {
            gaF += 8; gaH += 4; gw += 8;   // advance 32 elements
            if (A_BF16) { rh[0] = gaH[0]; rh[1] = gaH[1]; }
            else        { ra[0] = gaF[0]; ra[1] = gaF[1]; ra[2] = gaF[2]; ra[3] = gaF[3]; }
            rw[0] = gw[0]; rw[1] = gw[1]; rw[2] = gw[2]; rw[3] = gw[3];
        }

        // ---- fragments from LDS + WMMA ----
        FragU fa[2], fb[4];
        #pragma unroll
        for (int mt = 0; mt < 2; ++mt) {
            const uint4* p = (const uint4*)(curA + (waveM * 32 + mt * 16 + llo) * 24 + lhi * 8);
            fa[mt].q[0] = p[0]; fa[mt].q[1] = p[1];
        }
        #pragma unroll
        for (int nt = 0; nt < 4; ++nt) {
            const uint4* p = (const uint4*)(curB + (waveN * 64 + nt * 16 + llo) * 24 + lhi * 8);
            fb[nt].q[0] = p[0]; fb[nt].q[1] = p[1];
        }
        #pragma unroll
        for (int mt = 0; mt < 2; ++mt)
            #pragma unroll
            for (int nt = 0; nt < 4; ++nt)
                acc[mt][nt] = __builtin_amdgcn_wmma_f32_16x16x32_bf16(
                    false, fa[mt].v, false, fb[nt].v, (short)0, acc[mt][nt], false, false);

        // ---- stage next tile into the other buffer ----
        if (more) {
            unsigned* nA = ldsA + ((s + 1) & 1) * (128 * 24);
            unsigned* nB = ldsB + ((s + 1) & 1) * (128 * 24);
            if (A_BF16) {
                uint4* d = (uint4*)(nA + r * 24 + half * 8);
                d[0] = rh[0]; d[1] = rh[1];
            } else {
                unsigned* d = nA + r * 24 + half * 8;
                d[0] = pk2(ra[0].x, ra[0].y); d[1] = pk2(ra[0].z, ra[0].w);
                d[2] = pk2(ra[1].x, ra[1].y); d[3] = pk2(ra[1].z, ra[1].w);
                d[4] = pk2(ra[2].x, ra[2].y); d[5] = pk2(ra[2].z, ra[2].w);
                d[6] = pk2(ra[3].x, ra[3].y); d[7] = pk2(ra[3].z, ra[3].w);
            }
            unsigned* d = nB + r * 24 + half * 8;
            d[0] = pk2(rw[0].x, rw[0].y); d[1] = pk2(rw[0].z, rw[0].w);
            d[2] = pk2(rw[1].x, rw[1].y); d[3] = pk2(rw[1].z, rw[1].w);
            d[4] = pk2(rw[2].x, rw[2].y); d[5] = pk2(rw[2].z, rw[2].w);
            d[6] = pk2(rw[3].x, rw[3].y); d[7] = pk2(rw[3].z, rw[3].w);
        }
        __syncthreads();
    }

    // ---- epilogue: C layout lane=(N=llo, M=v+8*lhi) ----
    #pragma unroll
    for (int mt = 0; mt < 2; ++mt)
        #pragma unroll
        for (int nt = 0; nt < 4; ++nt) {
            size_t mb = (size_t)(m0 + waveM * 32 + mt * 16 + 8 * lhi) * N;
            int    n  = n0 + waveN * 64 + nt * 16 + llo;
            #pragma unroll
            for (int v = 0; v < 8; ++v) {
                float val = acc[mt][nt][v];
                if (OUT_BF16)
                    ((unsigned short*)Cp)[mb + (size_t)v * N + n] = f2bf(val);
                else
                    ((float*)Cp)[mb + (size_t)v * N + n] = val;
            }
        }
}

// ---------------------------------------------------------------------------
// RoPE on interleaved pairs, bf16 in-place. One thread per rotation pair:
// single b32 load, packed b32 store.
// ---------------------------------------------------------------------------
template<int NH>
__global__ __launch_bounds__(256) void rope_bf16(
    unsigned short* __restrict__ X, const float* __restrict__ Ct,
    const float* __restrict__ St)
{
    constexpr int S = 2048;
    size_t idx = (size_t)blockIdx.x * 256 + threadIdx.x;
    int    j  = (int)(idx & 63);
    size_t rr = idx >> 6;
    int    h  = (int)(rr % NH);
    size_t bs = rr / NH;              // b*S + s
    int    s  = (int)(bs & (S - 1));
    size_t off = bs * (size_t)NH * 128 + (size_t)h * 128 + 2 * j;
    unsigned xv = *(const unsigned*)(X + off);
    float xr = bf2f((unsigned short)xv);
    float xi = bf2f((unsigned short)(xv >> 16));
    float c = Ct[s * 64 + j], sn = St[s * 64 + j];
    *(unsigned*)(X + off) = pk2(xr * c - xi * sn, xr * sn + xi * c);
}

// ---------------------------------------------------------------------------
// Causal flash attention. Block = 128 thr (4 waves), 64 q-rows per block,
// 16 q-rows per wave. K-tiles of 32 keys. Q*K^T and P*V via bf16 WMMA.
// K tile fetched by the Tensor Data Mover (wave 0) when available.
// k-loop split: mask-free bulk phase + 2 boundary tiles with compiled-in mask.
// ---------------------------------------------------------------------------
__global__ __launch_bounds__(128) void flash_attn_bf16(
    const unsigned short* __restrict__ Q,   // [B*S, H*128] roped bf16
    const unsigned short* __restrict__ Kb,  // [B*S, KVH*128] roped bf16
    const unsigned short* __restrict__ Vb,  // [B*S, KVH*128] bf16
    unsigned short* __restrict__ O)         // [B*S, H*128] bf16
{
    constexpr int S = 2048, H = 32, KVH = 8, HD = 128;
    constexpr int strideQ = H * HD;     // 4096
    constexpr int strideKV = KVH * HD;  // 1024

    __shared__ unsigned ldsK[32 * 68];      // 32 keys x (128+8) bf16, row-major
    __shared__ unsigned ldsV[128 * 20];     // V^T: 128 cols x (32+8) bf16
    __shared__ unsigned ldsP[4 * 16 * 20];  // per-wave P: 16 x (32+8) bf16

    const int tid  = threadIdx.x;
    const int lane = tid & 31, wave = tid >> 5;
    const int llo  = lane & 15, lhi = lane >> 4;
    const int q0 = blockIdx.x * 64;
    const int h  = blockIdx.y, b = blockIdx.z;
    const int kvh = h >> 2;                 // H/KVH = 4
    const int bS = b * S;
    const float scale = 0.08838834764831845f;   // 128^-0.5

    // Q fragments: 16 rows x 128, split into 4 K-steps of 32
    v16bf qf[4];
    {
        size_t qrow = (size_t)(bS + q0 + wave * 16 + llo);
        #pragma unroll
        for (int j = 0; j < 4; ++j) {
            const uint4* p = (const uint4*)(Q + qrow * strideQ + h * HD + j * 32 + lhi * 16);
            FragU f; f.q[0] = p[0]; f.q[1] = p[1]; qf[j] = f.v;
        }
    }

    v8f vzero = {};
    v8f oacc[8];
    #pragma unroll
    for (int i = 0; i < 8; ++i) oacc[i] = vzero;
    float mi[8], li[8];
    #pragma unroll
    for (int v = 0; v < 8; ++v) { mi[v] = -3.0e38f; li[v] = 0.0f; }

    unsigned short* Pw = (unsigned short*)ldsP + wave * 16 * 40;

    auto tile = [&](int kb, auto maskC) {
        constexpr bool MASK = decltype(maskC)::value;
        // ---- K tile: 32 rows x 128 bf16 into padded LDS rows (272B stride) ----
#if HAVE_TDM
        if (wave == 0) {
            unsigned ldsOff = (unsigned)(unsigned long long)(void*)&ldsK[0];
            unsigned long long ga =
                (unsigned long long)(Kb + (size_t)(bS + kb) * strideKV + kvh * HD);
            v4u g0; v8i g1; v4i g2; v4i g3; v8i g4;
            g0.x = 1u;                                  // count=1 valid descriptor
            g0.y = ldsOff;                              // lds_addr
            g0.z = (unsigned)ga;                        // global_addr[31:0]
            g0.w = (unsigned)(ga >> 32) | (2u << 30);   // global_addr[56:32] | type=2
            // data_size=8B, pad_enable, pad_interval=256B, pad_amount=16B
            g1[0] = (int)((3u << 16) | (1u << 20) | (5u << 22) | (3u << 25));
            g1[1] = 0;                                  // tensor_dim0 = 0x40000000 (lo16=0)
            g1[2] = 0x4000;                             // tensor_dim0 hi16 | tensor_dim1 lo16
            g1[3] = (int)(0x4000u | (32u << 16));       // tensor_dim1 hi16 | tile_dim0=32 qw
            g1[4] = 32;                                 // tile_dim1=32 rows, tile_dim2=0
            g1[5] = 256;                                // tensor_dim0_stride = 256 qw (2048B)
            g1[6] = 0; g1[7] = 0;
            g2[0] = g2[1] = g2[2] = g2[3] = 0;
            g3[0] = g3[1] = g3[2] = g3[3] = 0;
            g4[0] = g4[1] = g4[2] = g4[3] = 0;
            g4[4] = g4[5] = g4[6] = g4[7] = 0;
            __builtin_amdgcn_tensor_load_to_lds(g0, g1, g2, g3, g4, 0);
            __builtin_amdgcn_s_wait_tensorcnt(0);
        }
#else
        {
            int r = tid >> 2, seg = tid & 3;
            const uint4* g = (const uint4*)(Kb + (size_t)(bS + kb + r) * strideKV +
                                            kvh * HD + seg * 32);
            uint4* d = (uint4*)(ldsK + r * 68 + seg * 16);
            d[0] = g[0]; d[1] = g[1]; d[2] = g[2]; d[3] = g[3];
        }
#endif
        // ---- V tile transposed: thread owns column c; constant strides ->
        //      immediate-offset global loads off a single base pointer ----
        {
            int c = tid;
            const unsigned short* g = Vb + (size_t)(bS + kb) * strideKV + kvh * HD + c;
            #pragma unroll
            for (int rp = 0; rp < 16; ++rp) {
                unsigned lo = g[(2 * rp) * strideKV];
                unsigned hi = g[(2 * rp + 1) * strideKV];
                ldsV[c * 20 + rp] = lo | (hi << 16);
            }
        }
        __syncthreads();

        // ---- scores: 16 q-rows x 32 keys ----
        v8f s0 = vzero, s1 = vzero;
        #pragma unroll
        for (int j = 0; j < 4; ++j) {
            FragU k0, k1;
            const uint4* p0 = (const uint4*)(ldsK + (llo)      * 68 + j * 16 + lhi * 8);
            const uint4* p1 = (const uint4*)(ldsK + (16 + llo) * 68 + j * 16 + lhi * 8);
            k0.q[0] = p0[0]; k0.q[1] = p0[1];
            k1.q[0] = p1[0]; k1.q[1] = p1[1];
            s0 = __builtin_amdgcn_wmma_f32_16x16x32_bf16(false, qf[j], false, k0.v,
                                                         (short)0, s0, false, false);
            s1 = __builtin_amdgcn_wmma_f32_16x16x32_bf16(false, qf[j], false, k1.v,
                                                         (short)0, s1, false, false);
        }

        // ---- online softmax (rows spread over 16-lane groups) ----
        float alpha[8];
        #pragma unroll
        for (int v = 0; v < 8; ++v) {
            float a0 = s0[v] * scale;
            float a1 = s1[v] * scale;
            if (MASK) {
                const int qrow = q0 + wave * 16 + v + 8 * lhi;
                a0 = ((kb + llo)      > qrow) ? -1.0e9f : a0;
                a1 = ((kb + 16 + llo) > qrow) ? -1.0e9f : a1;
            }
            float t = fmaxf(a0, a1);
            t = fmaxf(t, __shfl_xor(t, 1));
            t = fmaxf(t, __shfl_xor(t, 2));
            t = fmaxf(t, __shfl_xor(t, 4));
            t = fmaxf(t, __shfl_xor(t, 8));
            float mn = fmaxf(mi[v], t);
            float al = __expf(mi[v] - mn);
            float p0 = __expf(a0 - mn);
            float p1 = __expf(a1 - mn);
            float rs = p0 + p1;
            rs += __shfl_xor(rs, 1);
            rs += __shfl_xor(rs, 2);
            rs += __shfl_xor(rs, 4);
            rs += __shfl_xor(rs, 8);
            li[v] = li[v] * al + rs;
            mi[v] = mn;
            alpha[v] = al;
            Pw[(v + 8 * lhi) * 40 + llo]      = f2bf(p0);   // C-layout -> LDS
            Pw[(v + 8 * lhi) * 40 + 16 + llo] = f2bf(p1);
        }
        #pragma unroll
        for (int t8 = 0; t8 < 8; ++t8)
            #pragma unroll
            for (int v = 0; v < 8; ++v)
                oacc[t8][v] *= alpha[v];

        __syncthreads();   // order P b16 stores vs uint4 reload

        // ---- P (A-layout) and P*V ----
        FragU pf;
        {
            const uint4* p = (const uint4*)((unsigned*)ldsP + wave * 320 + llo * 20 + lhi * 8);
            pf.q[0] = p[0]; pf.q[1] = p[1];
        }
        #pragma unroll
        for (int t8 = 0; t8 < 8; ++t8) {
            FragU vf;
            const uint4* p = (const uint4*)(ldsV + (t8 * 16 + llo) * 20 + lhi * 8);
            vf.q[0] = p[0]; vf.q[1] = p[1];
            oacc[t8] = __builtin_amdgcn_wmma_f32_16x16x32_bf16(false, pf.v, false, vf.v,
                                                               (short)0, oacc[t8], false, false);
        }
        __syncthreads();   // protect K/V tiles before next iteration
    };

    // ---- phase 1: tiles fully below the diagonal, no masking code at all ----
    for (int kb = 0; kb < q0; kb += 32)
        tile(kb, BoolC<false>{});
    // ---- phase 2: the two diagonal-crossing tiles (kb = q0, q0+32) ----
    for (int kb = q0; kb <= q0 + 63; kb += 32)
        tile(kb, BoolC<true>{});

    // ---- normalize + store bf16 ----
    float inv[8];
    #pragma unroll
    for (int v = 0; v < 8; ++v) inv[v] = 1.0f / li[v];
    const size_t orow = (size_t)(bS + q0 + wave * 16);
    #pragma unroll
    for (int t8 = 0; t8 < 8; ++t8)
        #pragma unroll
        for (int v = 0; v < 8; ++v) {
            size_t m = orow + v + 8 * lhi;
            int    n = h * HD + t8 * 16 + llo;
            O[m * strideQ + n] = f2bf(oacc[t8][v] * inv[v]);
        }
}

// ---------------------------------------------------------------------------
extern "C" void kernel_launch(void* const* d_in, const int* in_sizes, int n_in,
                              void* d_out, int out_size, void* d_ws, size_t ws_size,
                              hipStream_t stream)
{
    (void)in_sizes; (void)n_in; (void)out_size; (void)ws_size;
    const float* x  = (const float*)d_in[0];
    const float* fc = (const float*)d_in[1];
    const float* fs = (const float*)d_in[2];
    // d_in[3] positions, d_in[4] mask: folded into kernels analytically
    const float* wq = (const float*)d_in[5];
    const float* wk = (const float*)d_in[6];
    const float* wv = (const float*)d_in[7];
    const float* wo = (const float*)d_in[8];
    float* out = (float*)d_out;

    const int B = 2, S = 2048, D = 4096, H = 32, KVH = 8, HD = 128;
    const int M   = B * S;       // 4096
    const int NQ  = H * HD;      // 4096
    const int NKV = KVH * HD;    // 1024

    unsigned short* qbf = (unsigned short*)d_ws;
    unsigned short* kbf = qbf + (size_t)M * NQ;
    unsigned short* vbf = kbf + (size_t)M * NKV;
    unsigned short* abf = vbf + (size_t)M * NKV;   // total 80 MB of ws

    dim3 blk(256);
    // QKV projections (fp32 A, bf16 out)
    gemm_nt_bf16<false, true><<<dim3(NQ / 128,  M / 128), blk, 0, stream>>>(x, wq, qbf, M, NQ,  D);
    gemm_nt_bf16<false, true><<<dim3(NKV / 128, M / 128), blk, 0, stream>>>(x, wk, kbf, M, NKV, D);
    gemm_nt_bf16<false, true><<<dim3(NKV / 128, M / 128), blk, 0, stream>>>(x, wv, vbf, M, NKV, D);
    // RoPE on Q and K
    rope_bf16<32><<<dim3(B * S * H   * 64 / 256), blk, 0, stream>>>(qbf, fc, fs);
    rope_bf16<8> <<<dim3(B * S * KVH * 64 / 256), blk, 0, stream>>>(kbf, fc, fs);
    // causal flash attention
    flash_attn_bf16<<<dim3(S / 64, H, B), dim3(128), 0, stream>>>(qbf, kbf, vbf, abf);
    // output projection (bf16 A, fp32 out)
    gemm_nt_bf16<true, false><<<dim3(D / 128, M / 128), blk, 0, stream>>>(abf, wo, out, M, D, NQ);
}
